// NerfNet_44229573214903
// MI455X (gfx1250) — matrix operand order repaired and checked by reading, not compiled, not run on hardware
//
#include <hip/hip_runtime.h>
#include <math.h>

// ---------------------------------------------------------------------------
// NeRF++ forward pass for MI455X (gfx1250, wave32, WMMA).
// Dominant cost: ~155 GFLOP of 256-wide GEMMs over 131072 points, weights
// ~2.4MB (L2-resident) -> compute bound -> V_WMMA_F32_16X16X32_F16 with f32
// accumulation, f16 operands staged in LDS.
// ---------------------------------------------------------------------------

typedef __attribute__((ext_vector_type(16))) _Float16 v16h;
typedef __attribute__((ext_vector_type(8)))  _Float16 v8h;
typedef __attribute__((ext_vector_type(8)))  float    v8f;

#define N_RAYS    2048
#define N_SAMPLES 64
#define N_PTS     (N_RAYS * N_SAMPLES)   // 131072
#define ROWS_BLK  128                    // points per block in the MLP kernel

// ---- workspace layout (float element offsets) ----
#define WS_Z      0                       // [131072] rescaled z_vals
#define WS_FAR    (N_PTS)                 // [2048]   fg_far_depth
#define WS_SIG    (WS_FAR + N_RAYS)       // [131072] sigma
#define WS_RGB    (WS_SIG + N_PTS)        // [131072*3] rgb
#define WS_WTS_F  (WS_RGB + 3 * N_PTS)    // f16 weights start here (657408 floats in)

// ---- f16 weight half-element offsets within the weight arena ----
#define OFF_W0    0          // [256][64]
#define OFF_W1    16384      // [256][256]
#define OFF_W2    81920
#define OFF_W3    147456
#define OFF_W4    212992     // [256][320]  (skip layer, padded col 63)
#define OFF_W5    294912
#define OFF_W6    360448
#define OFF_W7    425984
#define OFF_REMAP 491520     // [256][256]
#define OFF_G1    557056     // [128][288]
#define OFF_G2    593920     // [16][128]

// ---- d_out layout (flat concat in reference return order) ----
#define OUT_RGBMAP 0         // [2048*3]
#define OUT_DEPTH  6144      // [2048]
#define OUT_W      8192      // [131072]
#define OUT_ALPHA  139264    // [131072]
#define OUT_Z      270336    // [131072]
#define OUT_FAR    401408    // [2048]

// ===========================================================================
// Weight prep: f32 (fan_in, fan_out) row-major -> f16 Wt[Npad][Kpad] (B^T
// layout so B-fragments read two runs of 8 contiguous K values per lane). pad_at >= 0
// inserts a zero K-row at that index (skip-layer padding 319 -> 320).
// ===========================================================================
__global__ void __launch_bounds__(256)
prep_wt_kernel(const float* __restrict__ Wm, int fan_in, int fan_out,
               _Float16* __restrict__ out, int Npad, int Kpad, int pad_at)
{
    int idx = blockIdx.x * blockDim.x + threadIdx.x;
    if (idx >= Npad * Kpad) return;
    int n = idx / Kpad, k = idx % Kpad;
    int ks = k;
    bool valid = true;
    if (pad_at >= 0) {
        if (k == pad_at) valid = false;
        else if (k > pad_at) ks = k - 1;
    }
    float v = 0.0f;
    if (valid && n < fan_out && ks < fan_in) v = Wm[ks * fan_out + n];
    out[idx] = (_Float16)v;
}

// ===========================================================================
// FarDepth MLP: one wave per ray. hidden(256) = 8 regs/lane (lane + 32*o),
// shfl-broadcast inner products. Writes rescaled z_vals + far depth.
// ===========================================================================
struct FdPtrs {
    const float* ray_o; const float* ray_d; const float* near_d; const float* z_in;
    const float* W[5];  const float* b[5];
    float* ws_z; float* ws_far; float* out_z; float* out_far;
};

__global__ void __launch_bounds__(256) fd_kernel(FdPtrs P)
{
    const int lane = threadIdx.x & 31;
    const int wave = threadIdx.x >> 5;
    const int ray  = blockIdx.x * 8 + wave;

    float x[6];
#pragma unroll
    for (int c = 0; c < 3; ++c) { x[c] = P.ray_o[ray*3+c]; x[3+c] = P.ray_d[ray*3+c]; }

    float h[8];
#pragma unroll
    for (int o = 0; o < 8; ++o) {
        int j = lane + 32*o;
        float acc = P.b[0][j];
#pragma unroll
        for (int i = 0; i < 6; ++i) acc += x[i] * P.W[0][i*256 + j];
        h[o] = fmaxf(acc, 0.0f);
    }
    for (int L = 1; L < 4; ++L) {
        float nh[8];
#pragma unroll
        for (int o = 0; o < 8; ++o) nh[o] = P.b[L][lane + 32*o];
        for (int q = 0; q < 8; ++q) {
            for (int li = 0; li < 32; ++li) {
                float hv = __shfl(h[q], li, 32);
                const float* wrow = P.W[L] + (q*32 + li) * 256 + lane;
#pragma unroll
                for (int o = 0; o < 8; ++o) nh[o] += hv * wrow[32*o];
            }
        }
#pragma unroll
        for (int o = 0; o < 8; ++o) h[o] = fmaxf(nh[o], 0.0f);
    }
    float part = 0.0f;
#pragma unroll
    for (int o = 0; o < 8; ++o) part += h[o] * P.W[4][lane + 32*o];
#pragma unroll
    for (int off = 16; off >= 1; off >>= 1) part += __shfl_xor(part, off, 32);

    float fd  = 1.0f / (1.0f + expf(-(part + P.b[4][0])));
    float far = fd * 9.0f + 1.0f;
    float nr  = P.near_d[ray];
    for (int s = lane; s < 64; s += 32) {
        float zv = P.z_in[ray*64 + s];
        float z  = nr * (1.0f - zv) + far * zv;
        P.ws_z[ray*64 + s]  = z;
        P.out_z[ray*64 + s] = z;
    }
    if (lane == 0) { P.ws_far[ray] = far; P.out_far[ray] = far; }
}

// ===========================================================================
// WMMA fragment loaders (CDNA5 16x16x32 f16 layouts, wave32).
// A (16x32): lane m=l&15, half=l>>4 -> 8 contiguous f16 at K = half*8 and
//            8 contiguous f16 at K = 16 + half*8  (two 16B ds/global loads).
// B (32x16): stored as Wt[N][K]; lane n=l&15, half=l>>4 -> 8 contiguous f16
//            at K = half*8 and 8 at K = 16 + half*8 (two 16B loads).
// ===========================================================================
union Frag { v16h v; v8h h[2]; };

__device__ __forceinline__ v16h load_a_frag(const _Float16* base, int ld, int lane)
{
    const _Float16* p = base + (lane & 15) * ld + (lane >> 4) * 8;
    Frag a; a.h[0] = *(const v8h*)p; a.h[1] = *(const v8h*)(p + 16);
    return a.v;
}
__device__ __forceinline__ v16h load_b_frag(const _Float16* base, int ld, int lane)
{
    const _Float16* p = base + (lane & 15) * ld + (lane >> 4) * 8;
    Frag b; b.h[0] = *(const v8h*)p; b.h[1] = *(const v8h*)(p + 16);
    return b.v;
}

// out[128][N] = act(A[128][K] @ Wt^T + bias). A may be split in K: first K0
// columns from A0 (stride ld0), remainder from A1 (stride ld1); K0 % 32 == 0.
// Each wave owns N-strips of 16 (B fragment reused across all 8 M tiles).
__device__ __forceinline__ void mlp_gemm(
    const _Float16* A0, int ld0, int K0,
    const _Float16* A1, int ld1,
    const _Float16* Wt, int K,
    const float* bias, int N,
    _Float16* out, int ldo, bool relu)
{
    const int lane = threadIdx.x & 31;
    const int wave = threadIdx.x >> 5;
    const int ntiles = N >> 4;
    for (int nt = wave; nt < ntiles; nt += 8) {
        v8f acc[8] = {};
        const int ktiles = K >> 5;
        for (int kt = 0; kt < ktiles; ++kt) {
            const int kk = kt << 5;
            v16h bf = load_b_frag(Wt + nt * 16 * K + kk, K, lane);
            if (kk < K0) {
#pragma unroll
                for (int mt = 0; mt < 8; ++mt) {
                    v16h af = load_a_frag(A0 + (mt * 16) * ld0 + kk, ld0, lane);
                    acc[mt] = __builtin_amdgcn_wmma_f32_16x16x32_f16(
                        false, af, false, bf, (short)0, acc[mt], false, false);
                }
            } else {
#pragma unroll
                for (int mt = 0; mt < 8; ++mt) {
                    v16h af = load_a_frag(A1 + (mt * 16) * ld1 + (kk - K0), ld1, lane);
                    acc[mt] = __builtin_amdgcn_wmma_f32_16x16x32_f16(
                        false, af, false, bf, (short)0, acc[mt], false, false);
                }
            }
        }
        // C/D layout: VGPR r -> row mt*16 + (l>>4)*8 + r, col nt*16 + (l&15)
        const int half = lane >> 4;
        const int ncol = nt * 16 + (lane & 15);
        const float bv = bias[ncol];
#pragma unroll
        for (int mt = 0; mt < 8; ++mt) {
#pragma unroll
            for (int r = 0; r < 8; ++r) {
                float v = acc[mt][r] + bv;
                if (relu) v = fmaxf(v, 0.0f);
                out[(mt * 16 + half * 8 + r) * ldo + ncol] = (_Float16)v;
            }
        }
    }
}

// ===========================================================================
// Fused MLPNet kernel: 128 points/block, embeddings + 8 base layers + heads.
// LDS: pemb 128x64 (16KB) | demb 128x32 (8KB) | buf0/buf1 128x256 (64KB each)
// = 152KB dynamic LDS (CDNA5 WGP has 320KB).
// ===========================================================================
struct NetPtrs {
    const float* ray_o; const float* ray_d;
    const float* zvals; const _Float16* wts;
    const float* bb[8];
    const float* remap_b; const float* g1_b; const float* g2_b;
    const float* sigma_W; const float* sigma_b;
    float* sig_out; float* rgb_out;
};

#define MLP_SMEM_BYTES ((128*64 + 128*32 + 2*128*256) * 2)  // 155648 B

__global__ void __launch_bounds__(256) mlpnet_kernel(NetPtrs P)
{
    extern __shared__ __align__(16) _Float16 smem[];
    _Float16* pemb = smem;               // 128 x 64  (col 63 zero-padded)
    _Float16* demb = pemb + 128 * 64;    // 128 x 32  (cols 27..31 zero)
    _Float16* buf0 = demb + 128 * 32;    // 128 x 256
    _Float16* buf1 = buf0 + 128 * 256;   // 128 x 256

    const int t   = threadIdx.x;
    const int blk = blockIdx.x;

    // ---- positional / directional embeddings into LDS ----
    {
        int row = t & 127;
        int g   = blk * 128 + row;
        int ray = g >> 6;
        float rd0 = P.ray_d[ray*3+0], rd1 = P.ray_d[ray*3+1], rd2 = P.ray_d[ray*3+2];
        if (t < 128) {
            float z  = P.zvals[g];
            float pv[3] = { P.ray_o[ray*3+0] + z*rd0,
                            P.ray_o[ray*3+1] + z*rd1,
                            P.ray_o[ray*3+2] + z*rd2 };
            _Float16* d = pemb + row * 64;
            d[0] = (_Float16)pv[0]; d[1] = (_Float16)pv[1]; d[2] = (_Float16)pv[2];
#pragma unroll
            for (int f = 0; f < 10; ++f) {
                float fr = (float)(1 << f);
#pragma unroll
                for (int c = 0; c < 3; ++c) {
                    float a = pv[c] * fr;
                    d[3 + 6*f + c]     = (_Float16)sinf(a);
                    d[3 + 6*f + 3 + c] = (_Float16)cosf(a);
                }
            }
            d[63] = (_Float16)0.0f;
        } else {
            float inv = 1.0f / (sqrtf(rd0*rd0 + rd1*rd1 + rd2*rd2) + 1e-4f);
            float pv[3] = { rd0*inv, rd1*inv, rd2*inv };
            _Float16* d = demb + row * 32;
            d[0] = (_Float16)pv[0]; d[1] = (_Float16)pv[1]; d[2] = (_Float16)pv[2];
#pragma unroll
            for (int f = 0; f < 4; ++f) {
                float fr = (float)(1 << f);
#pragma unroll
                for (int c = 0; c < 3; ++c) {
                    float a = pv[c] * fr;
                    d[3 + 6*f + c]     = (_Float16)sinf(a);
                    d[3 + 6*f + 3 + c] = (_Float16)cosf(a);
                }
            }
#pragma unroll
            for (int k = 27; k < 32; ++k) d[k] = (_Float16)0.0f;
        }
    }
    __syncthreads();

    const _Float16* W = P.wts;
    __builtin_prefetch(W + OFF_W1, 0, 1);   // global_prefetch next-layer weights

    // ---- 8 base layers (skip-concat at layer 4) ----
    mlp_gemm(pemb, 64, 64,  (const _Float16*)0, 0, W + OFF_W0, 64,  P.bb[0], 256, buf0, 256, true);  __syncthreads();
    mlp_gemm(buf0, 256, 256,(const _Float16*)0, 0, W + OFF_W1, 256, P.bb[1], 256, buf1, 256, true);  __syncthreads();
    mlp_gemm(buf1, 256, 256,(const _Float16*)0, 0, W + OFF_W2, 256, P.bb[2], 256, buf0, 256, true);  __syncthreads();
    mlp_gemm(buf0, 256, 256,(const _Float16*)0, 0, W + OFF_W3, 256, P.bb[3], 256, buf1, 256, true);  __syncthreads();
    mlp_gemm(pemb, 64, 64,  buf1, 256,             W + OFF_W4, 320, P.bb[4], 256, buf0, 256, true);  __syncthreads();
    mlp_gemm(buf0, 256, 256,(const _Float16*)0, 0, W + OFF_W5, 256, P.bb[5], 256, buf1, 256, true);  __syncthreads();
    mlp_gemm(buf1, 256, 256,(const _Float16*)0, 0, W + OFF_W6, 256, P.bb[6], 256, buf0, 256, true);  __syncthreads();
    mlp_gemm(buf0, 256, 256,(const _Float16*)0, 0, W + OFF_W7, 256, P.bb[7], 256, buf1, 256, true);  __syncthreads();

    // ---- sigma head: |h @ sigma_W + b| (VALU dot, h_final in buf1) ----
    if (t < 128) {
        float acc = P.sigma_b[0];
        const _Float16* hr = buf1 + t * 256;
        for (int i = 0; i < 256; ++i) acc += (float)hr[i] * P.sigma_W[i];
        P.sig_out[blk * 128 + t] = fabsf(acc);
    }
    // ---- remap: base = h @ remap_W + b (reads buf1, writes buf0) ----
    mlp_gemm(buf1, 256, 256, (const _Float16*)0, 0, W + OFF_REMAP, 256, P.remap_b, 256, buf0, 256, false);
    __syncthreads();
    // ---- rgb layer 1: relu([base(256), demb(27pad32)] @ g1W) -> buf1[128x128]
    mlp_gemm(buf0, 256, 256, demb, 32, W + OFF_G1, 288, P.g1_b, 128, buf1, 128, true);
    __syncthreads();
    // ---- rgb layer 2: sigmoid(buf1 @ g2W), N padded 3->16, one M-tile/wave --
    {
        const int lane = t & 31;
        const int wave = t >> 5;
        v8f acc = {};
#pragma unroll
        for (int kt = 0; kt < 4; ++kt) {
            v16h bf = load_b_frag(W + OFF_G2 + kt * 32, 128, lane);
            v16h af = load_a_frag(buf1 + (wave * 16) * 128 + kt * 32, 128, lane);
            acc = __builtin_amdgcn_wmma_f32_16x16x32_f16(
                false, af, false, bf, (short)0, acc, false, false);
        }
        const int n = lane & 15, half = lane >> 4;
        if (n < 3) {
            float bv = P.g2_b[n];
#pragma unroll
            for (int r = 0; r < 8; ++r) {
                int m = wave * 16 + half * 8 + r;
                float v = acc[r] + bv;
                P.rgb_out[(blk * 128 + m) * 3 + n] = 1.0f / (1.0f + expf(-v));
            }
        }
    }
}

// ===========================================================================
// Compositing: one thread per ray, 64-step transmittance scan.
// ===========================================================================
__global__ void __launch_bounds__(256)
composite_kernel(const float* __restrict__ ray_d, const float* __restrict__ wsf,
                 float* __restrict__ out)
{
    int r = blockIdx.x * blockDim.x + threadIdx.x;
    if (r >= N_RAYS) return;
    float d0 = ray_d[r*3], d1 = ray_d[r*3+1], d2 = ray_d[r*3+2];
    float dn  = sqrtf(d0*d0 + d1*d1 + d2*d2);
    float far = wsf[WS_FAR + r];
    const float* z   = wsf + WS_Z   + r * 64;
    const float* sg  = wsf + WS_SIG + r * 64;
    const float* rgb = wsf + WS_RGB + r * 64 * 3;

    float T = 1.0f, ra = 0.f, ga = 0.f, ba = 0.f, da = 0.f;
    float zcur = z[0];
    for (int s = 0; s < 64; ++s) {
        float znext = (s < 63) ? z[s + 1] : 0.0f;
        float dist  = ((s < 63) ? (znext - zcur) : far) * dn;
        float alpha = 1.0f - expf(-sg[s] * dist);
        float w     = alpha * T;
        T *= (1.0f - alpha + 1e-4f);
        out[OUT_W     + r*64 + s] = w;
        out[OUT_ALPHA + r*64 + s] = alpha;
        ra += w * rgb[s*3+0]; ga += w * rgb[s*3+1]; ba += w * rgb[s*3+2];
        da += w * zcur;
        zcur = znext;
    }
    out[OUT_RGBMAP + r*3+0] = ra;
    out[OUT_RGBMAP + r*3+1] = ga;
    out[OUT_RGBMAP + r*3+2] = ba;
    out[OUT_DEPTH  + r]     = da;
}

// ===========================================================================
// Host launcher. Input order assumes pytree flattening of setup_inputs():
//  0 ray_o  1 ray_d  2 fg_near_depth  3 fg_far_depth0(unused)  4 fg_z_vals
//  5..9 fd Ws  10..14 fd bs
//  15..22 base_Ws  23..30 base_bs  31 remap_W  32 remap_b
//  33 rgb_Ws[0]  34 rgb_Ws[1]  35 rgb_bs[0]  36 rgb_bs[1]
//  37 sigma_W  38 sigma_b  39 idx  40 is_train
// ===========================================================================
extern "C" void kernel_launch(void* const* d_in, const int* in_sizes, int n_in,
                              void* d_out, int out_size, void* d_ws, size_t ws_size,
                              hipStream_t stream)
{
    (void)in_sizes; (void)n_in; (void)out_size; (void)ws_size;

    const float* ray_o = (const float*)d_in[0];
    const float* ray_d = (const float*)d_in[1];
    const float* neard = (const float*)d_in[2];
    const float* z_in  = (const float*)d_in[4];

    float*     wsf = (float*)d_ws;
    _Float16*  wts = (_Float16*)(wsf + WS_WTS_F);
    float*     out = (float*)d_out;

    // ---- weight prep (f32 -> transposed/padded f16) ----
    auto prep = [&](const float* Wm, int fin, int fout, int halfoff,
                    int Npad, int Kpad, int pad_at) {
        int total = Npad * Kpad;
        prep_wt_kernel<<<(total + 255) / 256, 256, 0, stream>>>(
            Wm, fin, fout, wts + halfoff, Npad, Kpad, pad_at);
    };
    const float* baseW[8]; const float* baseb[8];
    for (int i = 0; i < 8; ++i) { baseW[i] = (const float*)d_in[15 + i];
                                  baseb[i] = (const float*)d_in[23 + i]; }
    prep(baseW[0], 63, 256, OFF_W0, 256,  64, -1);
    prep(baseW[1], 256,256, OFF_W1, 256, 256, -1);
    prep(baseW[2], 256,256, OFF_W2, 256, 256, -1);
    prep(baseW[3], 256,256, OFF_W3, 256, 256, -1);
    prep(baseW[4], 319,256, OFF_W4, 256, 320, 63);   // skip concat: zero K-row at 63
    prep(baseW[5], 256,256, OFF_W5, 256, 256, -1);
    prep(baseW[6], 256,256, OFF_W6, 256, 256, -1);
    prep(baseW[7], 256,256, OFF_W7, 256, 256, -1);
    prep((const float*)d_in[31], 256, 256, OFF_REMAP, 256, 256, -1);
    prep((const float*)d_in[33], 283, 128, OFF_G1,    128, 288, -1);
    prep((const float*)d_in[34], 128,   3, OFF_G2,     16, 128, -1);

    // ---- FarDepth MLP + z rescale (wave per ray) ----
    FdPtrs FP;
    FP.ray_o = ray_o; FP.ray_d = ray_d; FP.near_d = neard; FP.z_in = z_in;
    for (int i = 0; i < 5; ++i) { FP.W[i] = (const float*)d_in[5 + i];
                                  FP.b[i] = (const float*)d_in[10 + i]; }
    FP.ws_z   = wsf + WS_Z;    FP.ws_far  = wsf + WS_FAR;
    FP.out_z  = out + OUT_Z;   FP.out_far = out + OUT_FAR;
    fd_kernel<<<N_RAYS / 8, 256, 0, stream>>>(FP);

    // ---- fused WMMA MLP over all 131072 points ----
    NetPtrs NP;
    NP.ray_o = ray_o; NP.ray_d = ray_d;
    NP.zvals = wsf + WS_Z; NP.wts = wts;
    for (int i = 0; i < 8; ++i) NP.bb[i] = baseb[i];
    NP.remap_b = (const float*)d_in[32];
    NP.g1_b    = (const float*)d_in[35];
    NP.g2_b    = (const float*)d_in[36];
    NP.sigma_W = (const float*)d_in[37];
    NP.sigma_b = (const float*)d_in[38];
    NP.sig_out = wsf + WS_SIG;
    NP.rgb_out = wsf + WS_RGB;
    mlpnet_kernel<<<N_PTS / ROWS_BLK, 256, MLP_SMEM_BYTES, stream>>>(NP);

    // ---- per-ray compositing ----
    composite_kernel<<<N_RAYS / 256, 256, 0, stream>>>(ray_d, wsf, out);
}
